// FusedMoE_25572235280544
// MI455X (gfx1250) — compile-verified, matrix operand order checked by reading, not run on hardware
//
#include <hip/hip_runtime.h>
#include <hip/hip_bf16.h>

#define T_TOK 4096
#define HDIM  1024
#define IDIM  2816
#define NEXP  8

typedef __attribute__((ext_vector_type(16))) __bf16 v16bf;
typedef __attribute__((ext_vector_type(8)))  float  v8f;

struct __align__(16) U4 { unsigned int x, y, z, w; };
union Frag { v16bf v; U4 q[2]; };

__device__ inline unsigned short f2bf(float f) {
    unsigned u = __builtin_bit_cast(unsigned, f);
    unsigned r = u + 0x7FFFu + ((u >> 16) & 1u);
    return (unsigned short)(r >> 16);
}

// CDNA5 async global->LDS copy (16B per lane), tracked by ASYNCcnt.
__device__ inline void async_cp16(const void* gsrc, const void* ldst) {
    unsigned lds_off = (unsigned)(unsigned long long)ldst;  // addr[31:0] == LDS byte offset
    asm volatile("global_load_async_to_lds_b128 %0, %1, off"
                 :: "v"(lds_off), "v"(gsrc) : "memory");
}

// ---------------- routing: softmax over 8 experts, top-2, build gather lists
__global__ void route_kernel(const float* __restrict__ logits,
                             int* __restrict__ counts,
                             int* __restrict__ etok,
                             float* __restrict__ escale,
                             int* __restrict__ tpair) {
    int t = blockIdx.x * blockDim.x + threadIdx.x;
    if (t >= T_TOK) return;
    float p[NEXP];
    float mx = -1e30f;
    #pragma unroll
    for (int i = 0; i < NEXP; ++i) { p[i] = logits[t * NEXP + i]; mx = fmaxf(mx, p[i]); }
    float s = 0.f;
    #pragma unroll
    for (int i = 0; i < NEXP; ++i) { p[i] = __expf(p[i] - mx); s += p[i]; }
    float inv = 1.0f / s;
    int i0 = 0;
    #pragma unroll
    for (int i = 1; i < NEXP; ++i) if (p[i] > p[i0]) i0 = i;
    int i1 = (i0 == 0) ? 1 : 0;
    #pragma unroll
    for (int i = 0; i < NEXP; ++i) if (i != i0 && p[i] > p[i1]) i1 = i;
    int   sel[2] = { i0, i1 };
    float sv[2]  = { p[i0] * inv, p[i1] * inv };
    #pragma unroll
    for (int k = 0; k < 2; ++k) {
        int e = sel[k];
        int pos = atomicAdd(&counts[e], 1);
        etok[e * T_TOK + pos]   = t;
        escale[e * T_TOK + pos] = sv[k];
        tpair[2 * t + k] = (e << 16) | pos;
    }
}

__global__ void scan_kernel(const int* __restrict__ counts, int* __restrict__ offs) {
    if (threadIdx.x == 0 && blockIdx.x == 0) {
        int acc = 0;
        for (int e = 0; e < NEXP; ++e) { offs[e] = acc; acc += counts[e]; }
        offs[NEXP] = acc;
    }
}

// ---------------- fp32 -> bf16 conversion (grid-strided)
__global__ void cvt_bf16(const float* __restrict__ src, unsigned short* __restrict__ dst, long n) {
    long i = (long)blockIdx.x * blockDim.x + threadIdx.x;
    long stride = (long)gridDim.x * blockDim.x;
    for (; i < n; i += stride) dst[i] = f2bf(src[i]);
}

// ---------------- GEMM1 + SwiGLU: h = silu(x@w1^T) * (x@w3^T), gathered rows
// BM=128, BN=64; 8 waves as 4(M)x2(N), each wave computes 32x32 per matrix.
__global__ __launch_bounds__(256) void gemm1_swiglu(
    const unsigned short* __restrict__ xb,      // [T,H] bf16
    const unsigned short* __restrict__ wb,      // [E,2I,H] bf16 (rows[:I]=w3 up, rows[I:]=w1 gate)
    const int* __restrict__ etok,
    const int* __restrict__ counts,
    const int* __restrict__ offs,
    unsigned short* __restrict__ hbuf) {        // [2T, I] bf16 compacted
    const int e = blockIdx.z;
    const int cnt = counts[e];
    const int row0 = blockIdx.y * 128;
    if (row0 >= cnt) return;
    const int n0 = blockIdx.x * 64;
    const int tid = threadIdx.x;

    __shared__ __align__(16) unsigned short sA[2][128][72];
    __shared__ __align__(16) unsigned short sBu[2][64][72];
    __shared__ __align__(16) unsigned short sBg[2][64][72];
    __shared__ int sTok[128];

    if (tid < 128) {
        int g = row0 + tid;
        sTok[tid] = (g < cnt) ? etok[e * T_TOK + g] : -1;
    }
    __syncthreads();

    const int wid = tid >> 5, lane = tid & 31;
    const int wm = wid & 3, wn = wid >> 2;              // 4 x 2 waves
    const int nl = lane & 15, hi = lane >> 4, hi8 = hi * 8;

    // A staging: 2 threads/row, 4 chunks each (row = tid>>1, base = (tid&1)*32)
    const int arow_ld = tid >> 1;
    const int abase   = (tid & 1) * 32;
    // B staging: 4 threads/row, 2 chunks each (row = tid>>2, offsets c0e/c1e)
    const int brow_ld = tid >> 2;
    const int c0e = (tid & 3) * 8;
    const int c1e = c0e + 32;

    const size_t wbase = (size_t)e * 2 * IDIM * HDIM;
    const int tok = sTok[arow_ld];
    const int tokc = tok < 0 ? 0 : tok;                 // clamp (garbage rows never stored)
    const unsigned short* gA  = &xb[(size_t)tokc * HDIM];
    const unsigned short* gBu = &wb[wbase + (size_t)(n0 + brow_ld) * HDIM];
    const unsigned short* gBg = &wb[wbase + (size_t)(n0 + brow_ld + IDIM) * HDIM];

    v8f aU[2][2] = {{{}, {}}, {{}, {}}};
    v8f aG[2][2] = {{{}, {}}, {{}, {}}};

    auto issue = [&](int k0, int b) {
        #pragma unroll
        for (int j = 0; j < 4; ++j)
            async_cp16(gA + k0 + abase + j * 8, &sA[b][arow_ld][abase + j * 8]);
        async_cp16(gBu + k0 + c0e, &sBu[b][brow_ld][c0e]);
        async_cp16(gBu + k0 + c1e, &sBu[b][brow_ld][c1e]);
        async_cp16(gBg + k0 + c0e, &sBg[b][brow_ld][c0e]);
        async_cp16(gBg + k0 + c1e, &sBg[b][brow_ld][c1e]);
    };

    issue(0, 0);
    const int nK = HDIM / 64;
    for (int ks = 0; ks < nK; ++ks) {
        const int b = ks & 1;
        if (ks + 1 < nK) {
            issue((ks + 1) * 64, b ^ 1);
            asm volatile("s_wait_asynccnt 0x8" ::: "memory");
        } else {
            asm volatile("s_wait_asynccnt 0x0" ::: "memory");
        }
        __syncthreads();
        #pragma unroll
        for (int kk = 0; kk < 2; ++kk) {
            const int ko = kk * 32;
            Frag a[2], bu[2], bg[2];
            #pragma unroll
            for (int ms = 0; ms < 2; ++ms) {
                int ar = wm * 32 + ms * 16 + nl;
                a[ms].q[0] = *((const U4*)&sA[b][ar][ko + hi8]);
                a[ms].q[1] = *((const U4*)&sA[b][ar][ko + 16 + hi8]);
            }
            #pragma unroll
            for (int ns = 0; ns < 2; ++ns) {
                int cc = wn * 32 + ns * 16 + nl;
                bu[ns].q[0] = *((const U4*)&sBu[b][cc][ko + hi8]);
                bu[ns].q[1] = *((const U4*)&sBu[b][cc][ko + 16 + hi8]);
                bg[ns].q[0] = *((const U4*)&sBg[b][cc][ko + hi8]);
                bg[ns].q[1] = *((const U4*)&sBg[b][cc][ko + 16 + hi8]);
            }
            #pragma unroll
            for (int ms = 0; ms < 2; ++ms) {
                #pragma unroll
                for (int ns = 0; ns < 2; ++ns) {
                    aU[ms][ns] = __builtin_amdgcn_wmma_f32_16x16x32_bf16(
                        false, a[ms].v, false, bu[ns].v, (short)0, aU[ms][ns], false, false);
                    aG[ms][ns] = __builtin_amdgcn_wmma_f32_16x16x32_bf16(
                        false, a[ms].v, false, bg[ns].v, (short)0, aG[ms][ns], false, false);
                }
            }
        }
        __syncthreads();
    }

    const int off = offs[e];
    #pragma unroll
    for (int ms = 0; ms < 2; ++ms) {
        #pragma unroll
        for (int r = 0; r < 8; ++r) {
            int grow = row0 + wm * 32 + ms * 16 + r + 8 * hi;
            if (grow < cnt) {
                size_t hb = (size_t)(off + grow) * IDIM;
                #pragma unroll
                for (int ns = 0; ns < 2; ++ns) {
                    int col = n0 + wn * 32 + ns * 16 + nl;
                    float u = aU[ms][ns][r], g = aG[ms][ns][r];
                    hbuf[hb + col] = f2bf((g / (1.0f + __expf(-g))) * u);
                }
            }
        }
    }
}

// ---------------- GEMM2 + routing scale: y = (h @ w2^T) * scale
// BM=128, BN=64; 8 waves as 4(M)x2(N), each wave 32x32.
__global__ __launch_bounds__(256) void gemm2_scale(
    const unsigned short* __restrict__ hbuf,    // [2T,I] bf16
    const unsigned short* __restrict__ w2b,     // [E,H,I] bf16
    const float* __restrict__ escale,           // [E,T]
    const int* __restrict__ counts,
    const int* __restrict__ offs,
    float* __restrict__ ybuf) {                 // [2T,H] f32
    const int e = blockIdx.z;
    const int cnt = counts[e];
    const int row0 = blockIdx.y * 128;
    if (row0 >= cnt) return;
    const int n0 = blockIdx.x * 64;
    const int tid = threadIdx.x;

    __shared__ __align__(16) unsigned short sA[2][128][72];
    __shared__ __align__(16) unsigned short sB[2][64][72];

    const int wid = tid >> 5, lane = tid & 31;
    const int wm = wid & 3, wn = wid >> 2;
    const int nl = lane & 15, hi = lane >> 4, hi8 = hi * 8;

    const int arow_ld = tid >> 1;
    const int abase   = (tid & 1) * 32;
    const int brow_ld = tid >> 2;
    const int c0e = (tid & 3) * 8;
    const int c1e = c0e + 32;

    const int off = offs[e];
    const size_t wbase = (size_t)e * HDIM * IDIM;

    int arow = off + row0 + arow_ld;
    if (arow > 2 * T_TOK - 1) arow = 2 * T_TOK - 1;     // clamp; garbage rows never stored
    const unsigned short* gA = &hbuf[(size_t)arow * IDIM];
    const unsigned short* gB = &w2b[wbase + (size_t)(n0 + brow_ld) * IDIM];

    v8f acc[2][2] = {{{}, {}}, {{}, {}}};

    auto issue = [&](int k0, int b) {
        #pragma unroll
        for (int j = 0; j < 4; ++j)
            async_cp16(gA + k0 + abase + j * 8, &sA[b][arow_ld][abase + j * 8]);
        async_cp16(gB + k0 + c0e, &sB[b][brow_ld][c0e]);
        async_cp16(gB + k0 + c1e, &sB[b][brow_ld][c1e]);
    };

    issue(0, 0);
    const int nK = IDIM / 64;
    for (int ks = 0; ks < nK; ++ks) {
        const int b = ks & 1;
        if (ks + 1 < nK) {
            issue((ks + 1) * 64, b ^ 1);
            asm volatile("s_wait_asynccnt 0x6" ::: "memory");
        } else {
            asm volatile("s_wait_asynccnt 0x0" ::: "memory");
        }
        __syncthreads();
        #pragma unroll
        for (int kk = 0; kk < 2; ++kk) {
            const int ko = kk * 32;
            Frag a[2], bb[2];
            #pragma unroll
            for (int ms = 0; ms < 2; ++ms) {
                int ar = wm * 32 + ms * 16 + nl;
                a[ms].q[0] = *((const U4*)&sA[b][ar][ko + hi8]);
                a[ms].q[1] = *((const U4*)&sA[b][ar][ko + 16 + hi8]);
            }
            #pragma unroll
            for (int ns = 0; ns < 2; ++ns) {
                int cc = wn * 32 + ns * 16 + nl;
                bb[ns].q[0] = *((const U4*)&sB[b][cc][ko + hi8]);
                bb[ns].q[1] = *((const U4*)&sB[b][cc][ko + 16 + hi8]);
            }
            #pragma unroll
            for (int ms = 0; ms < 2; ++ms) {
                #pragma unroll
                for (int ns = 0; ns < 2; ++ns) {
                    acc[ms][ns] = __builtin_amdgcn_wmma_f32_16x16x32_bf16(
                        false, a[ms].v, false, bb[ns].v, (short)0, acc[ms][ns], false, false);
                }
            }
        }
        __syncthreads();
    }

    #pragma unroll
    for (int ms = 0; ms < 2; ++ms) {
        #pragma unroll
        for (int r = 0; r < 8; ++r) {
            int grow = row0 + wm * 32 + ms * 16 + r + 8 * hi;
            if (grow < cnt) {
                float sc = escale[e * T_TOK + grow];
                size_t yb = (size_t)(off + grow) * HDIM;
                #pragma unroll
                for (int ns = 0; ns < 2; ++ns) {
                    int col = n0 + wn * 32 + ns * 16 + nl;
                    ybuf[yb + col] = acc[ms][ns][r] * sc;
                }
            }
        }
    }
}

// ---------------- combine: out[t] = y[slot0] + y[slot1]  (deterministic, no atomics)
__global__ void combine_kernel(const float* __restrict__ ybuf,
                               const int* __restrict__ tpair,
                               const int* __restrict__ offs,
                               float* __restrict__ out) {
    long i = (long)blockIdx.x * blockDim.x + threadIdx.x;
    if (i >= (long)T_TOK * HDIM) return;
    int t = (int)(i / HDIM), c = (int)(i % HDIM);
    int tp0 = tpair[2 * t], tp1 = tpair[2 * t + 1];
    int s0 = offs[tp0 >> 16] + (tp0 & 0xFFFF);
    int s1 = offs[tp1 >> 16] + (tp1 & 0xFFFF);
    out[i] = ybuf[(size_t)s0 * HDIM + c] + ybuf[(size_t)s1 * HDIM + c];
}

extern "C" void kernel_launch(void* const* d_in, const int* in_sizes, int n_in,
                              void* d_out, int out_size, void* d_ws, size_t ws_size,
                              hipStream_t stream) {
    const float* x      = (const float*)d_in[0];
    const float* logits = (const float*)d_in[1];
    const float* w3w1   = (const float*)d_in[2];
    const float* w2     = (const float*)d_in[3];
    float* out = (float*)d_out;

    char* ws = (char*)d_ws;
    size_t o = 0;
    auto alloc = [&](size_t bytes) { size_t p = o; o = (o + bytes + 255) & ~(size_t)255; return p; };

    const long nx    = (long)T_TOK * HDIM;
    const long nw31  = (long)NEXP * 2 * IDIM * HDIM;
    const long nw2   = (long)NEXP * HDIM * IDIM;
    const long nh    = (long)2 * T_TOK * IDIM;
    const long ny    = (long)2 * T_TOK * HDIM;

    unsigned short* xb    = (unsigned short*)(ws + alloc(nx * 2));
    unsigned short* w31b  = (unsigned short*)(ws + alloc(nw31 * 2));
    unsigned short* w2b   = (unsigned short*)(ws + alloc(nw2 * 2));
    unsigned short* hbuf  = (unsigned short*)(ws + alloc(nh * 2));
    float*          ybuf  = (float*)(ws + alloc(ny * 4));
    int*   counts = (int*)(ws + alloc(NEXP * 4));
    int*   offs   = (int*)(ws + alloc((NEXP + 1) * 4));
    int*   etok   = (int*)(ws + alloc((size_t)NEXP * T_TOK * 4));
    float* escale = (float*)(ws + alloc((size_t)NEXP * T_TOK * 4));
    int*   tpair  = (int*)(ws + alloc((size_t)2 * T_TOK * 4));

    hipMemsetAsync(counts, 0, NEXP * sizeof(int), stream);

    route_kernel<<<(T_TOK + 255) / 256, 256, 0, stream>>>(logits, counts, etok, escale, tpair);
    scan_kernel<<<1, 1, 0, stream>>>(counts, offs);

    cvt_bf16<<<4096, 256, 0, stream>>>(x, xb, nx);
    cvt_bf16<<<8192, 256, 0, stream>>>(w3w1, w31b, nw31);
    cvt_bf16<<<8192, 256, 0, stream>>>(w2, w2b, nw2);

    gemm1_swiglu<<<dim3(IDIM / 64, T_TOK / 128, NEXP), 256, 0, stream>>>(
        xb, w31b, etok, counts, offs, hbuf);
    gemm2_scale<<<dim3(HDIM / 64, T_TOK / 128, NEXP), 256, 0, stream>>>(
        hbuf, w2b, escale, counts, offs, ybuf);
    combine_kernel<<<(int)(((long)T_TOK * HDIM + 255) / 256), 256, 0, stream>>>(
        ybuf, tpair, offs, out);
}
